// SIGIR_A_G_21199958573628
// MI455X (gfx1250) — matrix operand order repaired and checked by reading, not compiled, not run on hardware
//
#include <hip/hip_runtime.h>
#include <hip/hip_bf16.h>

// ---------------------------------------------------------------------------
// Autoformer auto-correlation attention for MI455X (gfx1250, wave32, WMMA).
//   1) q/k/v projections: bf16 WMMA GEMM, tiles staged via
//      GLOBAL_LOAD_ASYNC_TO_LDS_B128 (ASYNCcnt), fp32->bf16 cvt during
//      fragment build (co-executes with WMMA)
//   2) channel-mean circular correlation via FFT (mean over c commutes w/ irfft)
//   3) top-7 lags + per-batch softmax
//   4) delay aggregation (7 circular shifts of v_proj)
//   5) output projection: bf16 WMMA GEMM -> fp32 d_out
// ---------------------------------------------------------------------------

#define BB    16
#define LL    2048
#define DMODEL 512
#define TOPK  7
#define NEL   ((size_t)BB * LL * DMODEL)   // 16,777,216 elements

typedef __attribute__((ext_vector_type(16))) __bf16 v16bf;
typedef __attribute__((ext_vector_type(8)))  float  v8f;

// ---- CDNA5 async copy: 16B global -> LDS per lane, tracked by ASYNCcnt -----
__device__ __forceinline__ void async_copy16(unsigned lds_byte, const void* gaddr) {
    asm volatile("global_load_async_to_lds_b128 %0, %1, off"
                 :: "v"(lds_byte), "v"((unsigned long long)(uintptr_t)gaddr)
                 : "memory");
}
__device__ __forceinline__ void wait_async0() {
    asm volatile("s_wait_asynccnt 0" ::: "memory");
}
// Low 32 bits of a generic shared pointer = LDS byte offset (ISA 10.2 aperture).
__device__ __forceinline__ unsigned lds_off(const void* p) {
    return (unsigned)(uintptr_t)p;
}

// ======================= WMMA GEMM: C = A(MxK) * W^T(NxK) + bias ============
// K = N = 512.  Macro tile 128x64, 8 waves (4x2), each wave does 32x32 = 4 WMMA
// tiles of 16x16, K-step 32 (bf16 16x16x32).  Tiles staged async in native
// precision; bf16 conversion happens in the fragment gather.
template <typename TIN, bool STORE_F32>
__global__ __launch_bounds__(256) void gemm_wmma(const TIN* __restrict__ A,
                                                 const float* __restrict__ W,
                                                 const float* __restrict__ bias,
                                                 void* __restrict__ Cout, int M) {
    __shared__ __attribute__((aligned(16))) TIN   As[128 * 32];  // M x K tile
    __shared__ __attribute__((aligned(16))) float Ws[64 * 32];   // N x K tile

    const int tid  = threadIdx.x;
    const int lane = tid & 31;
    const int wid  = tid >> 5;         // 0..7
    const int waveM = wid & 3;         // 4 waves along M
    const int waveN = wid >> 2;        // 2 waves along N
    const int M0 = blockIdx.y * 128;
    const int N0 = blockIdx.x * 64;

    const int laneN  = lane & 15;
    const int laneHi = (lane >> 4) & 1;    // 0: lanes 0-15, 1: lanes 16-31

    const unsigned asBase = lds_off(&As[0]);
    const unsigned wsBase = lds_off(&Ws[0]);

    constexpr int EPC = 16 / sizeof(TIN);   // elements per 16B chunk
    constexpr int CPR = 32 / EPC;           // chunks per 32-element row

    v8f acc[2][2] = {};

    for (int k0 = 0; k0 < DMODEL; k0 += 32) {
        // ---- stage A tile (async, native type) ----
        #pragma unroll
        for (int i = tid; i < 128 * CPR; i += 256) {
            int row = i / CPR, cc = (i % CPR) * EPC;
            async_copy16(asBase + (unsigned)((row * 32 + cc) * sizeof(TIN)),
                         &A[(size_t)(M0 + row) * DMODEL + k0 + cc]);
        }
        // ---- stage W tile (async, fp32) ----
        #pragma unroll
        for (int i = tid; i < 64 * 8; i += 256) {
            int row = i >> 3, cc = (i & 7) * 4;
            async_copy16(wsBase + (unsigned)((row * 32 + cc) * 4),
                         &W[(size_t)(N0 + row) * DMODEL + k0 + cc]);
        }
        wait_async0();
        __syncthreads();

        // Build fragments per ISA 7.12.2 layouts (convert to bf16 here).
        // A (16x32, 16-bit): lane holds row M = lane%16;
        //   elem j -> K = (j<8 ? j : j+8) + (lane<16 ? 0 : 8)
        v16bf afrag[2];
        #pragma unroll
        for (int mi = 0; mi < 2; ++mi) {
            int row = waveM * 32 + mi * 16 + laneN;
            #pragma unroll
            for (int j = 0; j < 16; ++j) {
                int kk = ((j < 8) ? j : j + 8) + (laneHi ? 8 : 0);
                afrag[mi][j] = (__bf16)((float)As[row * 32 + kk]);
            }
        }
        // B (32x16, 16-bit): lane holds col N = lane%16;
    //   elem j -> K = (j<8 ? j : j+8) + (lane<16 ? 0 : 8).  B[k][n] = W[n][k].
        v16bf bfrag[2];
        #pragma unroll
        for (int ni = 0; ni < 2; ++ni) {
            int col = waveN * 32 + ni * 16 + laneN;
            #pragma unroll
            for (int j = 0; j < 16; ++j) {
                int kk = ((j < 8) ? j : j + 8) + (laneHi ? 8 : 0);
                bfrag[ni][j] = (__bf16)Ws[col * 32 + kk];
            }
        }

        #pragma unroll
        for (int mi = 0; mi < 2; ++mi)
            #pragma unroll
            for (int ni = 0; ni < 2; ++ni)
                acc[mi][ni] = __builtin_amdgcn_wmma_f32_16x16x32_bf16(
                    false, afrag[mi], false, bfrag[ni],
                    (short)0, acc[mi][ni], false, false);

        __syncthreads();
    }

    // Epilogue: C/D layout — VGPR r, lane: N = lane%16, M = r + 8*(lane>=16)
    #pragma unroll
    for (int mi = 0; mi < 2; ++mi) {
        #pragma unroll
        for (int ni = 0; ni < 2; ++ni) {
            #pragma unroll
            for (int r = 0; r < 8; ++r) {
                int m = M0 + waveM * 32 + mi * 16 + r + laneHi * 8;
                int n = N0 + waveN * 32 + ni * 16 + laneN;
                float val = acc[mi][ni][r] + bias[n];
                if (STORE_F32)
                    ((float*)Cout)[(size_t)m * DMODEL + n] = val;
                else
                    ((__bf16*)Cout)[(size_t)m * DMODEL + n] = (__bf16)val;
            }
        }
    }
}

// ======================= Radix-2 2048-pt FFT in LDS (256 threads) ===========
__device__ __forceinline__ void fft2048(float2* s, int tid) {
    // bit reversal (11 bits)
    for (int i = tid; i < 2048; i += 256) {
        int r = (int)(__brev((unsigned)i) >> 21);
        if (i < r) { float2 t = s[i]; s[i] = s[r]; s[r] = t; }
    }
    __syncthreads();
    for (int st = 1; st <= 11; ++st) {
        const int half = 1 << (st - 1);
        const int m    = half << 1;
        for (int w = tid; w < 1024; w += 256) {
            int grp = w >> (st - 1);
            int pos = w & (half - 1);
            int i0 = grp * m + pos;
            int i1 = i0 + half;
            float ang = -6.283185307179586f * (float)pos / (float)m;
            float sw, cw;
            __sincosf(ang, &sw, &cw);
            float2 u = s[i0], v = s[i1];
            float2 t = make_float2(v.x * cw - v.y * sw, v.x * sw + v.y * cw);
            s[i0] = make_float2(u.x + t.x, u.y + t.y);
            s[i1] = make_float2(u.x - t.x, u.y - t.y);
        }
        __syncthreads();
    }
}

// ===== Forward FFTs of q,k; accumulate sum_c Q*conj(K) over 16 channels =====
// grid (B, 32): each block handles 16 channels of one batch.
__global__ __launch_bounds__(256) void fft_corr_accum(const __bf16* __restrict__ qp,
                                                      const __bf16* __restrict__ kp,
                                                      float2* __restrict__ Acc) {
    __shared__ float2 bq[2048];
    __shared__ float2 bk[2048];
    __shared__ float accR[1025];
    __shared__ float accI[1025];

    const int b = blockIdx.x, chunk = blockIdx.y, tid = threadIdx.x;
    for (int f = tid; f <= 1024; f += 256) { accR[f] = 0.f; accI[f] = 0.f; }
    __syncthreads();

    for (int ci = 0; ci < 16; ++ci) {
        const int c = chunk * 16 + ci;
        for (int t = tid; t < 2048; t += 256) {
            size_t base = ((size_t)b * LL + t) * DMODEL + c;
            bq[t] = make_float2((float)qp[base], 0.f);
            bk[t] = make_float2((float)kp[base], 0.f);
        }
        __syncthreads();
        fft2048(bq, tid);
        fft2048(bk, tid);
        for (int f = tid; f <= 1024; f += 256) {
            float2 Q = bq[f], K = bk[f];
            accR[f] += Q.x * K.x + Q.y * K.y;   // Re(Q*conj(K))
            accI[f] += Q.y * K.x - Q.x * K.y;   // Im(Q*conj(K))
        }
        __syncthreads();
    }
    for (int f = tid; f <= 1024; f += 256) {
        atomicAdd(&Acc[(size_t)b * 1025 + f].x, accR[f]);
        atomicAdd(&Acc[(size_t)b * 1025 + f].y, accI[f]);
    }
}

// ===== 16 inverse FFTs: mcorr[b,tau] = irfft(Acc[b]) / (L*C) =================
__global__ __launch_bounds__(256) void ifft_corr(const float2* __restrict__ Acc,
                                                 float* __restrict__ mcorr) {
    __shared__ float2 buf[2048];
    const int b = blockIdx.x, tid = threadIdx.x;
    // Y = conj(X); irfft = Re(fft(Y))/N
    for (int f = tid; f <= 1024; f += 256) {
        float2 x = Acc[(size_t)b * 1025 + f];
        buf[f] = make_float2(x.x, -x.y);
    }
    __syncthreads();
    // Hermitian extension of Y: Y[2048-f] = conj(Y[f])
    for (int f = tid + 1; f <= 1023; f += 256) {
        float2 y = buf[f];
        buf[2048 - f] = make_float2(y.x, -y.y);
    }
    __syncthreads();
    fft2048(buf, tid);
    const float scale = (1.0f / 2048.0f) * (1.0f / (float)DMODEL);
    for (int t = tid; t < 2048; t += 256)
        mcorr[(size_t)b * LL + t] = buf[t].x * scale;
}

// ===== top-7 lags of mean_b mcorr, then per-batch softmax weights ===========
__global__ __launch_bounds__(256) void topk_softmax(const float* __restrict__ mcorr,
                                                    int* __restrict__ idx_out,
                                                    float* __restrict__ w_out) {
    __shared__ float mean[2048];
    __shared__ float rmax[256];
    __shared__ int   ridx[256];
    __shared__ int   sidx[TOPK];
    const int tid = threadIdx.x;

    for (int t = tid; t < 2048; t += 256) {
        float s = 0.f;
        for (int b = 0; b < BB; ++b) s += mcorr[(size_t)b * LL + t];
        mean[t] = s;   // scale irrelevant for argmax
    }
    __syncthreads();

    for (int k = 0; k < TOPK; ++k) {
        float best = -3.0e38f; int bi = 0;
        for (int t = tid; t < 2048; t += 256)
            if (mean[t] > best) { best = mean[t]; bi = t; }
        rmax[tid] = best; ridx[tid] = bi;
        __syncthreads();
        for (int off = 128; off > 0; off >>= 1) {
            if (tid < off && rmax[tid + off] > rmax[tid]) {
                rmax[tid] = rmax[tid + off]; ridx[tid] = ridx[tid + off];
            }
            __syncthreads();
        }
        if (tid == 0) {
            sidx[k] = ridx[0];
            idx_out[k] = ridx[0];
            mean[ridx[0]] = -3.0e38f;
        }
        __syncthreads();
    }

    if (tid < BB) {
        float wv[TOPK], mx = -3.0e38f;
        for (int k = 0; k < TOPK; ++k) {
            wv[k] = mcorr[(size_t)tid * LL + sidx[k]];
            mx = fmaxf(mx, wv[k]);
        }
        float s = 0.f;
        for (int k = 0; k < TOPK; ++k) { wv[k] = __expf(wv[k] - mx); s += wv[k]; }
        float inv = 1.0f / s;
        for (int k = 0; k < TOPK; ++k) w_out[tid * TOPK + k] = wv[k] * inv;
    }
}

// ===== delay aggregation: agg[b,l,c] = sum_k w[b,k] * v[b,(l+lag_k)%L,c] ====
__global__ __launch_bounds__(256) void aggregate(const __bf16* __restrict__ vp,
                                                 const int* __restrict__ idx,
                                                 const float* __restrict__ w,
                                                 __bf16* __restrict__ agg) {
    __shared__ int   sidx[TOPK];
    __shared__ float sw[BB * TOPK];
    if (threadIdx.x < TOPK) sidx[threadIdx.x] = idx[threadIdx.x];
    if (threadIdx.x < BB * TOPK) sw[threadIdx.x] = w[threadIdx.x];
    __syncthreads();

    size_t i = (size_t)blockIdx.x * 256 + threadIdx.x;   // < B*L*C
    int c = (int)(i & (DMODEL - 1));
    size_t rest = i >> 9;
    int l = (int)(rest & (LL - 1));
    int b = (int)(rest >> 11);

    float s = 0.f;
    #pragma unroll
    for (int k = 0; k < TOPK; ++k) {
        int lk = (l + sidx[k]) & (LL - 1);
        s += sw[b * TOPK + k] * (float)vp[((size_t)b * LL + lk) * DMODEL + c];
    }
    agg[i] = (__bf16)s;
}

__global__ void zero_f32(float* p, size_t n) {
    size_t i = (size_t)blockIdx.x * 256 + threadIdx.x;
    if (i < n) p[i] = 0.f;
}

// ============================ launch ========================================
extern "C" void kernel_launch(void* const* d_in, const int* in_sizes, int n_in,
                              void* d_out, int out_size, void* d_ws, size_t ws_size,
                              hipStream_t stream) {
    const float* q  = (const float*)d_in[0];
    const float* kI = (const float*)d_in[1];
    const float* v  = (const float*)d_in[2];
    const float* Wq = (const float*)d_in[3];
    const float* bq = (const float*)d_in[4];
    const float* Wk = (const float*)d_in[5];
    const float* bk = (const float*)d_in[6];
    const float* Wv = (const float*)d_in[7];
    const float* bv = (const float*)d_in[8];
    const float* Wo = (const float*)d_in[9];
    const float* bo = (const float*)d_in[10];

    char* ws = (char*)d_ws;
    __bf16* qp  = (__bf16*)ws;                       // NEL bf16
    __bf16* kp  = qp + NEL;                          // NEL bf16
    __bf16* vp  = kp + NEL;                          // NEL bf16
    __bf16* agg = vp + NEL;                          // NEL bf16
    char*   p   = (char*)(agg + NEL);
    float2* Acc   = (float2*)p;                      // 16*1025 float2
    p += (size_t)BB * 1025 * sizeof(float2);
    float*  mcorr = (float*)p;                       // 16*2048 float
    p += (size_t)BB * LL * sizeof(float);
    int*    lagix = (int*)p;                         // 7 ints
    p += 64;
    float*  wgt   = (float*)p;                       // 16*7 floats

    const int M = BB * LL;                           // 32768
    dim3 gg(DMODEL / 64, M / 128);                   // (8, 256)

    // 1) projections
    gemm_wmma<float, false><<<gg, 256, 0, stream>>>(q,  Wq, bq, (void*)qp, M);
    gemm_wmma<float, false><<<gg, 256, 0, stream>>>(kI, Wk, bk, (void*)kp, M);
    gemm_wmma<float, false><<<gg, 256, 0, stream>>>(v,  Wv, bv, (void*)vp, M);

    // 2) frequency-domain correlation, mean over channels
    size_t accFloats = (size_t)BB * 1025 * 2;
    zero_f32<<<(unsigned)((accFloats + 255) / 256), 256, 0, stream>>>((float*)Acc, accFloats);
    fft_corr_accum<<<dim3(BB, DMODEL / 16), 256, 0, stream>>>(qp, kp, Acc);
    ifft_corr<<<BB, 256, 0, stream>>>(Acc, mcorr);

    // 3) lag selection + softmax
    topk_softmax<<<1, 256, 0, stream>>>(mcorr, lagix, wgt);

    // 4) delay aggregation
    aggregate<<<(unsigned)(NEL / 256), 256, 0, stream>>>(vp, lagix, wgt, agg);

    // 5) output projection -> fp32 d_out
    gemm_wmma<__bf16, true><<<gg, 256, 0, stream>>>(agg, Wo, bo, d_out, M);
}
